// ARqGPS_14774687498325
// MI455X (gfx1250) — matrix operand order repaired and checked by reading, not compiled, not run on hardware
//
#include <hip/hip_runtime.h>

typedef __attribute__((ext_vector_type(16))) _Float16 v16h;
typedef __attribute__((ext_vector_type(8)))  float    v8f;
typedef __attribute__((ext_vector_type(4)))  float    v4f;

#define L_DIM   256
#define M_DIM   128
#define B_DIM   1024
#define HALF_L  128

// ---------------------------------------------------------------------------
// Pre-pass: transpose epsilon (2, M, L) -> epsT[i][s][m] (f32) and epsH (f16).
// ---------------------------------------------------------------------------
__global__ void eps_transpose_kernel(const float* __restrict__ eps,
                                     float* __restrict__ epsT,
                                     _Float16* __restrict__ epsH) {
  int tid = blockIdx.x * blockDim.x + threadIdx.x;   // 0 .. 2*M*L-1
  int i   = tid & (L_DIM - 1);
  int sm  = tid >> 8;                                // s*M + m
  float v = eps[sm * L_DIM + i];
  int o   = i * (2 * M_DIM) + sm;                    // [i][s][m]
  epsT[o] = v;
  epsH[o] = (_Float16)v;
}

// ---------------------------------------------------------------------------
// Main scan: one wave per tile of 16 batch rows; software-pipelined by 1 step.
//   A (16x32 f16) = cache P (f32 master, cvt per step)
//   B (32x16 f16) = eps[:, :, i] (cols 0,1 = spin channels)
//   D (16x16 f32) row r = batch row, cols 0/1 = conditional log-amplitudes.
// ---------------------------------------------------------------------------
__global__ __launch_bounds__(32)
void arqgps_scan_kernel(const int* __restrict__ inputs,
                        const float* __restrict__ epsT,
                        const _Float16* __restrict__ epsH,
                        float* __restrict__ out) {
  __shared__ float lds[16 * 16];

  const int lane = threadIdx.x;          // 0..31
  const int bl   = lane & 15;            // batch row within tile (== A/D row M)
  const int hi   = lane >> 4;            // K-half carried by this lane
  const int b    = blockIdx.x * 16 + bl;

  // f32 master cache. A-layout slot t (chunk c): m = c*32 + 16*hi + t
  float P[4][16];
  #pragma unroll
  for (int c = 0; c < 4; ++c)
    #pragma unroll
    for (int t = 0; t < 16; ++t)
      P[c][t] = 1.0f;

  const int* inRow = inputs + b * L_DIM;
  const int sB = (bl < 2) ? bl : 0;      // B col N=bl carries channel s (>=2 ignored)
  const _Float16* eBbase = epsH + sB * M_DIM + hi * 16;

  // ---- pipeline prologue: B tile for i=0, input for i=0 -------------------
  v16h bv0 = *(const v16h*)(eBbase + 0);
  v16h bv1 = *(const v16h*)(eBbase + 32);
  v16h bv2 = *(const v16h*)(eBbase + 64);
  v16h bv3 = *(const v16h*)(eBbase + 96);
  int inp_cur = inRow[0];

  float acc = 0.0f;
  int n1 = 0;                            // up-spins among sites j < i

  for (int i = 0; i < L_DIM; ++i) {
    // ---- D = A(P) x B(eps_i), chained over 4 K-chunks --------------------
    v8f d = {0.f, 0.f, 0.f, 0.f, 0.f, 0.f, 0.f, 0.f};
    {
      v16h a;
      #pragma unroll
      for (int t = 0; t < 16; ++t) a[t] = (_Float16)P[0][t];
      d = __builtin_amdgcn_wmma_f32_16x16x32_f16(false, a, false, bv0, (short)0, d, false, false);
      #pragma unroll
      for (int t = 0; t < 16; ++t) a[t] = (_Float16)P[1][t];
      d = __builtin_amdgcn_wmma_f32_16x16x32_f16(false, a, false, bv1, (short)0, d, false, false);
      #pragma unroll
      for (int t = 0; t < 16; ++t) a[t] = (_Float16)P[2][t];
      d = __builtin_amdgcn_wmma_f32_16x16x32_f16(false, a, false, bv2, (short)0, d, false, false);
      #pragma unroll
      for (int t = 0; t < 16; ++t) a[t] = (_Float16)P[3][t];
      d = __builtin_amdgcn_wmma_f32_16x16x32_f16(false, a, false, bv3, (short)0, d, false, false);
    }

    // ---- prefetch for the NEXT iteration (addresses known now) -----------
    int i1 = (i + 1 < L_DIM) ? (i + 1) : (L_DIM - 1);   // clamped; tail reload harmless
    int inp_nxt = inRow[i1];
    const float* eU = epsT + i * (2 * M_DIM) + inp_cur * M_DIM;  // update P_i -> P_{i+1}
    v4f e0[4], e1[4], e2[4], e3[4];
    #pragma unroll
    for (int c = 0; c < 4; ++c) {
      int base = c * 32 + 16 * hi;
      e0[c] = *(const v4f*)(eU + base);
      e1[c] = *(const v4f*)(eU + base + 4);
      e2[c] = *(const v4f*)(eU + base + 8);
      e3[c] = *(const v4f*)(eU + base + 12);
    }
    const _Float16* eBn = eBbase + i1 * (2 * M_DIM);
    bv0 = *(const v16h*)(eBn + 0);
    bv1 = *(const v16h*)(eBn + 32);
    bv2 = *(const v16h*)(eBn + 64);
    bv3 = *(const v16h*)(eBn + 96);

    // ---- exchange D via LDS: row M = r + 8*hi, col N = bl ----------------
    __syncthreads();
    #pragma unroll
    for (int r = 0; r < 8; ++r)
      lds[(hi * 8 + r) * 16 + bl] = d[r];
    __syncthreads();
    float v0 = lds[bl * 16 + 0];
    float v1 = lds[bl * 16 + 1];

    // ---- mask + stable 2-way logsumexp (hardware exp2/log2) --------------
    int n0 = i - n1;
    float lp0 = (n0 >= HALF_L) ? -__builtin_inff() : v0;
    float lp1 = (n1 >= HALF_L) ? -__builtin_inff() : v1;
    float lps = inp_cur ? lp1 : lp0;                   // picked channel, never -inf
    float mx  = fmaxf(lp0, lp1);
    float mn  = fminf(lp0, lp1);
    float dl  = mn - mx;                               // <= 0, may be -inf
    // exp(2*dl) = 2^(2*log2e*dl); v_exp_f32(-inf)=0, v_log_f32(1)=0
    float t2  = __builtin_amdgcn_exp2f(2.8853900817779268f * dl);
    float l2  = __builtin_amdgcn_logf(1.0f + t2);
    acc += (lps - mx) - 0.34657359027997264f * l2;     // lps - 0.5*lse

    n1 += inp_cur;

    // ---- apply prefetched update: P_{i+1} = P_i * eps[inp_i, :, i] -------
    #pragma unroll
    for (int c = 0; c < 4; ++c)
      #pragma unroll
      for (int k = 0; k < 4; ++k) {
        P[c][k]      *= e0[c][k];
        P[c][4 + k]  *= e1[c][k];
        P[c][8 + k]  *= e2[c][k];
        P[c][12 + k] *= e3[c][k];
      }
    inp_cur = inp_nxt;
  }

  if (lane < 16) out[b] = acc;
}

// ---------------------------------------------------------------------------
extern "C" void kernel_launch(void* const* d_in, const int* in_sizes, int n_in,
                              void* d_out, int out_size, void* d_ws, size_t ws_size,
                              hipStream_t stream) {
  const int*   inputs = (const int*)d_in[0];     // (B, L) int32
  const float* eps    = (const float*)d_in[1];   // (2, M, L) f32
  float*       out    = (float*)d_out;           // (B,) f32

  float*    epsT = (float*)d_ws;                                     // 256 KB
  _Float16* epsH = (_Float16*)((char*)d_ws + 2 * M_DIM * L_DIM * 4); // 128 KB

  int total = 2 * M_DIM * L_DIM;
  eps_transpose_kernel<<<total / 256, 256, 0, stream>>>(eps, epsT, epsH);
  arqgps_scan_kernel<<<B_DIM / 16, 32, 0, stream>>>(inputs, epsT, epsH, out);
}
